// Attention_34754875359410
// MI455X (gfx1250) — matrix operand order repaired and checked by reading, not compiled, not run on hardware
//
#include <hip/hip_runtime.h>

typedef __attribute__((ext_vector_type(16))) __bf16 v16bf;
typedef __attribute__((ext_vector_type(8)))  float  v8f;

#define S_LEN 2048
#define HID_DIM 2048
#define NH 16
#define NKV 4
#define HD 128
#define GROUPS (NH / NKV)

union ABu { v16bf v; uint4 q[2]; };

static __device__ __forceinline__ unsigned short f2bf(float f) {
  union { float f; unsigned u; } v; v.f = f;
  unsigned u = v.u + 0x7FFFu + ((v.u >> 16) & 1u);   // round-to-nearest-even
  return (unsigned short)(u >> 16);
}

// ---------------------------------------------------------------- converts
__global__ void attn_cvt_bf16(const float* __restrict__ X,
                              unsigned short* __restrict__ Y, int n) {
  int i = blockIdx.x * blockDim.x + threadIdx.x;
  if (i < n) Y[i] = f2bf(X[i]);
}

// Pack W (K x N row-major f32) into WMMA B-fragment bf16 layout:
// tile (kt,nt) of 32x16 -> 512 elems, per-lane 16 contiguous elems.
// element (k,n): lane = (n&15) | ((k>>4)&1)<<4 ; elem = k&15
__global__ void attn_pack_w(const float* __restrict__ W,
                            unsigned short* __restrict__ Wf, int K, int N) {
  int i = blockIdx.x * blockDim.x + threadIdx.x;
  if (i >= K * N) return;
  int k = i / N, n = i - k * N;
  int kt = k >> 5, nt = n >> 4;
  int lane = (n & 15) | (((k >> 4) & 1) << 4);
  int e = k & 15;
  size_t idx = (((size_t)(kt * (N >> 4) + nt)) << 9) + ((size_t)lane << 4) + e;
  Wf[idx] = f2bf(W[(size_t)k * N + n]);
}

// ---------------------------------------------------------------- GEMM (bf16 WMMA)
// C[M,N] f32 row-major = A[M,K] bf16 row-major  x  Bf (fragment layout)
__global__ __launch_bounds__(256) void attn_gemm_bf16(
    const unsigned short* __restrict__ A,
    const unsigned short* __restrict__ Bf,
    float* __restrict__ C, int M, int N, int K) {
  const int lane = threadIdx.x & 31;
  const int wid  = threadIdx.x >> 5;
  const int m0 = blockIdx.y * 128 + (wid & 3) * 32;
  const int n0 = blockIdx.x * 64  + (wid >> 2) * 32;
  const int ntiles = N >> 4;

  v8f c00{}, c01{}, c10{}, c11{};
  const int lrow0 = m0 + (lane & 15);
  const int lrow1 = lrow0 + 16;
  const int kshift = (lane >> 4) << 4;   // 0 or 16

  for (int kt = 0; kt < (K >> 5); ++kt) {
    const int kbase = (kt << 5) + kshift;
    ABu a0, a1, b0, b1;
    const unsigned short* pa0 = A + (size_t)lrow0 * K + kbase;
    const unsigned short* pa1 = A + (size_t)lrow1 * K + kbase;
    a0.q[0] = *(const uint4*)(pa0);      a0.q[1] = *(const uint4*)(pa0 + 8);
    a1.q[0] = *(const uint4*)(pa1);      a1.q[1] = *(const uint4*)(pa1 + 8);
    const unsigned short* pb = Bf + (((size_t)(kt * ntiles + (n0 >> 4))) << 9) + (lane << 4);
    b0.q[0] = *(const uint4*)(pb);       b0.q[1] = *(const uint4*)(pb + 8);
    b1.q[0] = *(const uint4*)(pb + 512); b1.q[1] = *(const uint4*)(pb + 520);

    c00 = __builtin_amdgcn_wmma_f32_16x16x32_bf16(false, a0.v, false, b0.v, (short)0, c00, false, false);
    c01 = __builtin_amdgcn_wmma_f32_16x16x32_bf16(false, a0.v, false, b1.v, (short)0, c01, false, false);
    c10 = __builtin_amdgcn_wmma_f32_16x16x32_bf16(false, a1.v, false, b0.v, (short)0, c10, false, false);
    c11 = __builtin_amdgcn_wmma_f32_16x16x32_bf16(false, a1.v, false, b1.v, (short)0, c11, false, false);
  }

  const int col  = lane & 15;
  const int rofs = (lane >> 4) << 3;
  for (int r = 0; r < 8; ++r) {
    C[(size_t)(m0 + rofs + r) * N + n0 + col]           = c00[r];
    C[(size_t)(m0 + rofs + r) * N + n0 + 16 + col]      = c01[r];
    C[(size_t)(m0 + 16 + rofs + r) * N + n0 + col]      = c10[r];
    C[(size_t)(m0 + 16 + rofs + r) * N + n0 + 16 + col] = c11[r];
  }
}

// ---------------------------------------------------------------- RoPE
__global__ void attn_rope_q(const float* __restrict__ Q, const int* __restrict__ pos,
                            unsigned short* __restrict__ Qbf) {
  int i = blockIdx.x * blockDim.x + threadIdx.x;         // NH*S*64
  if (i >= NH * S_LEN * 64) return;
  int d = i & 63; int s = (i >> 6) & (S_LEN - 1); int h = i >> 17;
  float ang = (float)pos[s] * __powf(10000.0f, -(float)d / 64.0f);
  float sn, cs; __sincosf(ang, &sn, &cs);
  const float* q = Q + (size_t)s * (NH * HD) + h * HD;
  float x1 = q[d], x2 = q[d + 64];
  unsigned short* out = Qbf + ((size_t)h * S_LEN + s) * HD;
  out[d]      = f2bf(x1 * cs - x2 * sn);
  out[d + 64] = f2bf(x1 * sn + x2 * cs);
}

// K: rope + emit repeated f32 k-output + B-fragment layout of Kh^T (contraction = d, N = s)
__global__ void attn_rope_k(const float* __restrict__ Kx, const int* __restrict__ pos,
                            unsigned short* __restrict__ Kfrag, float* __restrict__ k_out) {
  int i = blockIdx.x * blockDim.x + threadIdx.x;         // NKV*S*64
  if (i >= NKV * S_LEN * 64) return;
  int d = i & 63; int s = (i >> 6) & (S_LEN - 1); int kv = i >> 17;
  float ang = (float)pos[s] * __powf(10000.0f, -(float)d / 64.0f);
  float sn, cs; __sincosf(ang, &sn, &cs);
  const float* kp = Kx + (size_t)s * (NKV * HD) + kv * HD;
  float x1 = kp[d], x2 = kp[d + 64];
  float o1 = x1 * cs - x2 * sn;
  float o2 = x1 * sn + x2 * cs;
  for (int g = 0; g < GROUPS; ++g) {
    float* ko = k_out + ((size_t)(kv * GROUPS + g) * S_LEN + s) * HD;
    ko[d] = o1; ko[d + 64] = o2;
  }
  auto put = [&](int dd, float val) {
    int dk = dd >> 5, st = s >> 4;
    int lane = (s & 15) | (((dd >> 4) & 1) << 4);
    size_t idx = (size_t)kv * HD * S_LEN +
                 (((size_t)(dk * (S_LEN >> 4) + st)) << 9) + ((size_t)lane << 4) + (dd & 15);
    Kfrag[idx] = f2bf(val);
  };
  put(d, o1); put(d + 64, o2);
}

// V: repeated f32 v-output + B-fragment layout of Vh (contraction = s, N = d)
__global__ void attn_pack_v(const float* __restrict__ Vx,
                            unsigned short* __restrict__ Vfrag, float* __restrict__ v_out) {
  int i = blockIdx.x * blockDim.x + threadIdx.x;         // NKV*S*HD
  if (i >= NKV * S_LEN * HD) return;
  int d = i & 127; int s = (i >> 7) & (S_LEN - 1); int kv = i >> 18;
  float val = Vx[(size_t)s * (NKV * HD) + kv * HD + d];
  for (int g = 0; g < GROUPS; ++g)
    v_out[((size_t)(kv * GROUPS + g) * S_LEN + s) * HD + d] = val;
  int kt = s >> 5, dt = d >> 4;
  int lane = (d & 15) | (((s >> 4) & 1) << 4);
  size_t idx = (size_t)kv * S_LEN * HD +
               (((size_t)(kt * (HD >> 4) + dt)) << 9) + ((size_t)lane << 4) + (s & 15);
  Vfrag[idx] = f2bf(val);
}

// ---------------------------------------------------------------- flash attention
// one wave per (head, 16-query block); online softmax; causal mask analytic
__global__ __launch_bounds__(128) void attn_flash(
    const unsigned short* __restrict__ Qbf,    // [NH][S][HD] bf16 row-major
    const unsigned short* __restrict__ Kfrag,  // [NKV] fragment tiles of K^T
    const unsigned short* __restrict__ Vfrag,  // [NKV] fragment tiles of V
    unsigned short* __restrict__ Obf) {        // [S][NH*HD] bf16 row-major
  __shared__ unsigned short lds[4][16 * 32];
  const int lane = threadIdx.x & 31;
  const int wid  = threadIdx.x >> 5;
  const int task = blockIdx.x * 4 + wid;
  const int h  = task >> 7;                     // S/16 = 128 query blocks per head
  const int m0 = (task & 127) << 4;
  const int kv = h >> 2;                        // GQA: h / GROUPS
  const float scale = 0.08838834764831845f;     // 1/sqrt(128)

  ABu qa[4];
  {
    const unsigned short* qrow = Qbf + ((size_t)h * S_LEN + m0 + (lane & 15)) * HD;
    int kshift = (lane >> 4) << 4;
    for (int dk = 0; dk < 4; ++dk) {
      const unsigned short* p = qrow + dk * 32 + kshift;
      qa[dk].q[0] = *(const uint4*)(p);
      qa[dk].q[1] = *(const uint4*)(p + 8);
    }
  }
  const unsigned short* Kf = Kfrag + (size_t)kv * HD * S_LEN;
  const unsigned short* Vf = Vfrag + (size_t)kv * S_LEN * HD;

  v8f o[8]; for (int t = 0; t < 8; ++t) o[t] = v8f{};
  float mrow[8], lrow[8];
  for (int r = 0; r < 8; ++r) { mrow[r] = -3.0e38f; lrow[r] = 0.0f; }

  const int nkb = (m0 + 47) >> 5;               // key blocks of 32 with key <= m0+15
  for (int kb32 = 0; kb32 < nkb; ++kb32) {
    const int kb = kb32 << 5;
    v8f s0{}, s1{};
    const int st0 = kb >> 4;
    for (int dk = 0; dk < 4; ++dk) {
      ABu kb0, kb1;
      const unsigned short* p0 = Kf + (((size_t)(dk * (S_LEN >> 4) + st0)) << 9) + (lane << 4);
      kb0.q[0] = *(const uint4*)(p0);       kb0.q[1] = *(const uint4*)(p0 + 8);
      kb1.q[0] = *(const uint4*)(p0 + 512); kb1.q[1] = *(const uint4*)(p0 + 520);
      s0 = __builtin_amdgcn_wmma_f32_16x16x32_bf16(false, qa[dk].v, false, kb0.v, (short)0, s0, false, false);
      s1 = __builtin_amdgcn_wmma_f32_16x16x32_bf16(false, qa[dk].v, false, kb1.v, (short)0, s1, false, false);
    }
    // scale + causal mask
    const int key0  = kb + (lane & 15);
    const int rbase = m0 + ((lane >> 4) << 3);
    for (int r = 0; r < 8; ++r) {
      int qr = rbase + r;
      float v0 = s0[r] * scale; if (key0 > qr)      v0 = -1.0e9f;
      float v1 = s1[r] * scale; if (key0 + 16 > qr) v1 = -1.0e9f;
      s0[r] = v0; s1[r] = v1;
    }
    // online softmax (row reductions across 16-lane halves)
    v8f alpha;
    for (int r = 0; r < 8; ++r) {
      float mx = fmaxf(s0[r], s1[r]);
      for (int off = 1; off < 16; off <<= 1) mx = fmaxf(mx, __shfl_xor(mx, off, 32));
      float mnew = fmaxf(mrow[r], mx);
      float a = __expf(mrow[r] - mnew);
      mrow[r] = mnew; alpha[r] = a;
      float p0 = __expf(s0[r] - mnew);
      float p1 = __expf(s1[r] - mnew);
      s0[r] = p0; s1[r] = p1;
      float sum = p0 + p1;
      for (int off = 1; off < 16; off <<= 1) sum += __shfl_xor(sum, off, 32);
      lrow[r] = lrow[r] * a + sum;
    }
    for (int t = 0; t < 8; ++t) o[t] *= alpha;

    // transpose P (C layout -> A layout) through this wave's LDS slice
    unsigned short* P = lds[wid];
    {
      int c = lane & 15, rofs = (lane >> 4) << 3;
      for (int r = 0; r < 8; ++r) {
        P[(r + rofs) * 32 + c]      = f2bf(s0[r]);
        P[(r + rofs) * 32 + 16 + c] = f2bf(s1[r]);
      }
    }
    ABu pa;
    {
      const unsigned short* pr = P + (lane & 15) * 32 + ((lane >> 4) << 4);
      pa.q[0] = *(const uint4*)(pr);
      pa.q[1] = *(const uint4*)(pr + 8);
    }
    // O += P @ V_block
    for (int dt = 0; dt < 8; ++dt) {
      ABu vb;
      const unsigned short* pv = Vf + (((size_t)(kb32 * 8 + dt)) << 9) + (lane << 4);
      vb.q[0] = *(const uint4*)(pv); vb.q[1] = *(const uint4*)(pv + 8);
      o[dt] = __builtin_amdgcn_wmma_f32_16x16x32_bf16(false, pa.v, false, vb.v, (short)0, o[dt], false, false);
    }
  }

  v8f rl;
  for (int r = 0; r < 8; ++r) rl[r] = 1.0f / lrow[r];
  const int col = lane & 15, rofs = (lane >> 4) << 3;
  for (int dt = 0; dt < 8; ++dt) {
    v8f ov = o[dt] * rl;
    unsigned short* op = Obf + (size_t)(m0 + rofs) * (NH * HD) + h * HD + dt * 16 + col;
    for (int r = 0; r < 8; ++r) op[(size_t)r * (NH * HD)] = f2bf(ov[r]);
  }
}

// ---------------------------------------------------------------- launch
extern "C" void kernel_launch(void* const* d_in, const int* in_sizes, int n_in,
                              void* d_out, int out_size, void* d_ws, size_t ws_size,
                              hipStream_t stream) {
  const float* hs  = (const float*)d_in[0];
  const int*   pos = (const int*)d_in[1];
  // d_in[2] = mask (causal applied analytically)
  const float* Wq = (const float*)d_in[3];
  const float* Wk = (const float*)d_in[4];
  const float* Wv = (const float*)d_in[5];
  const float* Wo = (const float*)d_in[6];

  char* ws = (char*)d_ws; size_t ofs = 0;
  auto alloc = [&](size_t bytes) { void* p = ws + ofs; ofs += (bytes + 255) & ~(size_t)255; return p; };
  unsigned short* Xbf  = (unsigned short*)alloc((size_t)S_LEN * HID_DIM * 2);
  unsigned short* Wqf  = (unsigned short*)alloc((size_t)HID_DIM * NH * HD * 2);
  unsigned short* Wkf  = (unsigned short*)alloc((size_t)HID_DIM * NKV * HD * 2);
  unsigned short* Wvf  = (unsigned short*)alloc((size_t)HID_DIM * NKV * HD * 2);
  unsigned short* Wof  = (unsigned short*)alloc((size_t)NH * HD * HID_DIM * 2);
  float*          Qf   = (float*)alloc((size_t)S_LEN * NH * HD * 4);
  float*          Kf32 = (float*)alloc((size_t)S_LEN * NKV * HD * 4);
  float*          Vf32 = (float*)alloc((size_t)S_LEN * NKV * HD * 4);
  unsigned short* Qbf  = (unsigned short*)alloc((size_t)NH * S_LEN * HD * 2);
  unsigned short* Kfr  = (unsigned short*)alloc((size_t)NKV * HD * S_LEN * 2);
  unsigned short* Vfr  = (unsigned short*)alloc((size_t)NKV * S_LEN * HD * 2);
  unsigned short* Obf  = (unsigned short*)alloc((size_t)S_LEN * NH * HD * 2);

  float* out   = (float*)d_out;
  float* k_out = out + (size_t)S_LEN * HID_DIM;
  float* v_out = k_out + (size_t)NH * S_LEN * HD;

  { int n = S_LEN * HID_DIM;  attn_cvt_bf16<<<(n + 255) / 256, 256, 0, stream>>>(hs, Xbf, n); }
  { int n = HID_DIM * NH * HD;  attn_pack_w<<<(n + 255) / 256, 256, 0, stream>>>(Wq, Wqf, HID_DIM, NH * HD); }
  { int n = HID_DIM * NKV * HD; attn_pack_w<<<(n + 255) / 256, 256, 0, stream>>>(Wk, Wkf, HID_DIM, NKV * HD); }
  { int n = HID_DIM * NKV * HD; attn_pack_w<<<(n + 255) / 256, 256, 0, stream>>>(Wv, Wvf, HID_DIM, NKV * HD); }
  { int n = NH * HD * HID_DIM;  attn_pack_w<<<(n + 255) / 256, 256, 0, stream>>>(Wo, Wof, NH * HD, HID_DIM); }

  dim3 blk(256);
  attn_gemm_bf16<<<dim3((NH * HD) / 64, S_LEN / 128), blk, 0, stream>>>(Xbf, Wqf, Qf, S_LEN, NH * HD, HID_DIM);
  attn_gemm_bf16<<<dim3((NKV * HD) / 64, S_LEN / 128), blk, 0, stream>>>(Xbf, Wkf, Kf32, S_LEN, NKV * HD, HID_DIM);
  attn_gemm_bf16<<<dim3((NKV * HD) / 64, S_LEN / 128), blk, 0, stream>>>(Xbf, Wvf, Vf32, S_LEN, NKV * HD, HID_DIM);

  { int n = NH * S_LEN * 64;  attn_rope_q<<<(n + 255) / 256, 256, 0, stream>>>(Qf, pos, Qbf); }
  { int n = NKV * S_LEN * 64; attn_rope_k<<<(n + 255) / 256, 256, 0, stream>>>(Kf32, pos, Kfr, k_out); }
  { int n = NKV * S_LEN * HD; attn_pack_v<<<(n + 255) / 256, 256, 0, stream>>>(Vf32, Vfr, v_out); }

  attn_flash<<<dim3(NH * (S_LEN / 16) / 4), dim3(128), 0, stream>>>(Qbf, Kfr, Vfr, Obf);

  attn_gemm_bf16<<<dim3(HID_DIM / 64, S_LEN / 128), blk, 0, stream>>>(Obf, Wof, out, S_LEN, HID_DIM, NH * HD);
}
